// SimpleMamba_41609643164078
// MI455X (gfx1250) — matrix operand order repaired and checked by reading, not compiled, not run on hardware
//
#include <hip/hip_runtime.h>

typedef __bf16 bf16_t;
typedef __attribute__((ext_vector_type(16))) __bf16 v16bf;
typedef __attribute__((ext_vector_type(8)))  __bf16 v8bf;
typedef __attribute__((ext_vector_type(8)))  float  v8f;

#define B_SZ     8
#define T_SZ     4096
#define DIN      1024
#define DMODEL   1024
#define NCHUNK   8
#define TCHUNK   (T_SZ / NCHUNK)   // 512 timesteps per chunk
#define BN       128               // columns per workgroup (8 waves x 16)
#define MT       64                // rows per register-resident block (4 wmma tiles)
#define KG       64                // k elements staged per LDS group (2 wmma k-steps)
#define NG       (DIN / KG)        // 16 k-groups
#define LDSK     72                // padded row stride (bank-conflict-free: 36-bank rot/row)

static __device__ __forceinline__ float softplus_f(float x) {
    return (x > 20.0f) ? x : log1pf(expf(x));
}

static __device__ __forceinline__ v8f zero_v8f() {
    v8f z;
#pragma unroll
    for (int i = 0; i < 8; ++i) z[i] = 0.0f;
    return z;
}

// async global->LDS copy, 16 bytes (tracked by ASYNCcnt)
static __device__ __forceinline__ void async_b128(unsigned lds_addr, unsigned long long gaddr) {
    asm volatile("global_load_async_to_lds_b128 %0, %1, off"
                 :: "v"(lds_addr), "v"(gaddr) : "memory");
}

// ---------------- prep: split f32 -> bf16 hi/lo ----------------
__global__ void split_x_kernel(const float* __restrict__ src,
                               bf16_t* __restrict__ hi, bf16_t* __restrict__ lo,
                               int n) {
    int i = blockIdx.x * blockDim.x + threadIdx.x;
    if (i < n) {
        float v = src[i];
        bf16_t h = (bf16_t)v;
        hi[i] = h;
        lo[i] = (bf16_t)(v - (float)h);
    }
}

// transpose W (k-major -> n-major) + split into bf16 hi/lo, all 3 matrices
__global__ void split_w_kernel(const float* __restrict__ W0,
                               const float* __restrict__ W1,
                               const float* __restrict__ W2,
                               bf16_t* __restrict__ wh, bf16_t* __restrict__ wl) {
    int i = blockIdx.x * blockDim.x + threadIdx.x;   // i = mat*2^20 + n*1024 + k
    if (i >= 3 * DIN * DMODEL) return;
    int mat = i >> 20;
    int rem = i & ((1 << 20) - 1);
    int n = rem >> 10;
    int k = rem & 1023;
    const float* W = (mat == 0) ? W0 : ((mat == 1) ? W1 : W2);
    float v = W[k * DMODEL + n];
    bf16_t h = (bf16_t)v;
    wh[i] = h;
    wl[i] = (bf16_t)(v - (float)h);
}

// ---------------- fused projections + gating + chunk-local affine scan ----------------
__global__ __launch_bounds__(256)
void mamba_proj_scan_kernel(const bf16_t* __restrict__ xh, const bf16_t* __restrict__ xl,
                            const bf16_t* __restrict__ wh, const bf16_t* __restrict__ wl,
                            const float* __restrict__ b_lam, const float* __restrict__ b_del,
                            const float* __restrict__ b_B,
                            float* __restrict__ Achunk, float* __restrict__ Dchunk) {
    // [buf][hi/lo][row][k] ; padded rows for conflict-free 16-row fragment reads
    __shared__ __align__(16) bf16_t sa[2][2][MT][LDSK];

    const int nblk  = blockIdx.x;           // column block (BN cols)
    const int b     = blockIdx.y;           // batch
    const int chunk = blockIdx.z;           // T chunk
    const int wave  = threadIdx.x >> 5;
    const int lane  = threadIdx.x & 31;
    const int lm    = lane & 15;
    const bool hiHalf = lane >= 16;

    const int n = nblk * BN + wave * 16 + lm;   // this lane's output channel

    const float blam = b_lam[n];
    const float bdel = b_del[n];
    const float bBv  = b_B[n];

    // A operand k order used here (same as B): lanes 0-15 hold K=0..15, lanes 16-31 hold K=16..31 (contig)
    const int kA = hiHalf ? 16 : 0;
    // B operand k order used here (same as A): lanes 0-15 hold K=0..15, lanes 16-31 hold K=16..31 (contig)
    const int kB = hiHalf ? 16 : 0;

    const bf16_t* whp[3];
    const bf16_t* wlp[3];
#pragma unroll
    for (int m = 0; m < 3; ++m) {
        size_t off = ((size_t)m << 20) + (size_t)n * DIN + kB;
        whp[m] = wh + off;
        wlp[m] = wl + off;
    }

    // cooperative staging indices: 4 threads per row, 16 elems (32B) each
    const int srow = threadIdx.x >> 2;          // 0..63
    const int skc  = (threadIdx.x & 3) * 16;    // elem offset within group

    float Acar = 1.0f, Dcar = 0.0f;   // chunk-local affine map s -> Acar*s + Dcar
    const int t0chunk = chunk * TCHUNK;

    for (int mb = 0; mb < TCHUNK / MT; ++mb) {
        v8f acc[4][3];
#pragma unroll
        for (int t = 0; t < 4; ++t)
#pragma unroll
            for (int m = 0; m < 3; ++m) acc[t][m] = zero_v8f();

        const size_t rowg = (size_t)b * T_SZ + (size_t)(t0chunk + mb * MT + srow);
        const unsigned long long gH = (unsigned long long)(size_t)(xh + rowg * DIN + skc);
        const unsigned long long gL = (unsigned long long)(size_t)(xl + rowg * DIN + skc);

        // initial prefetch: group 0 -> buf 0 (4 async b128 per thread)
        {
            unsigned lH = (unsigned)(unsigned long long)(size_t)&sa[0][0][srow][skc];
            unsigned lL = (unsigned)(unsigned long long)(size_t)&sa[0][1][srow][skc];
            async_b128(lH,      gH);
            async_b128(lH + 16, gH + 16);
            async_b128(lL,      gL);
            async_b128(lL + 16, gL + 16);
        }

        for (int g = 0; g < NG; ++g) {
            const int buf = g & 1;
            if (g + 1 < NG) {
                // prefetch next group into the other buffer
                unsigned lH = (unsigned)(unsigned long long)(size_t)&sa[buf ^ 1][0][srow][skc];
                unsigned lL = (unsigned)(unsigned long long)(size_t)&sa[buf ^ 1][1][srow][skc];
                unsigned long long goff = (unsigned long long)((g + 1) * KG) * sizeof(bf16_t);
                async_b128(lH,      gH + goff);
                async_b128(lH + 16, gH + goff + 16);
                async_b128(lL,      gL + goff);
                async_b128(lL + 16, gL + goff + 16);
                asm volatile("s_wait_asynccnt 0x4" ::: "memory");  // current group done
            } else {
                asm volatile("s_wait_asynccnt 0x0" ::: "memory");
            }
            __syncthreads();   // current buffer visible to all waves

#pragma unroll
            for (int ks = 0; ks < KG / 32; ++ks) {
                const int kk = g * KG + ks * 32;   // global k for B fragments
                const int kl = ks * 32;            // local k inside LDS group
                v16bf bhf[3], blf[3];
#pragma unroll
                for (int m = 0; m < 3; ++m) {
                    bhf[m] = *(const v16bf*)(whp[m] + kk);
                    blf[m] = *(const v16bf*)(wlp[m] + kk);
                }
#pragma unroll
                for (int t = 0; t < 4; ++t) {
                    const int row = t * 16 + lm;
                    union { v16bf v; v8bf h[2]; } ah, al;
                    ah.h[0] = *(const v8bf*)&sa[buf][0][row][kA + kl];
                    ah.h[1] = *(const v8bf*)&sa[buf][0][row][kA + kl + 8];
                    al.h[0] = *(const v8bf*)&sa[buf][1][row][kA + kl];
                    al.h[1] = *(const v8bf*)&sa[buf][1][row][kA + kl + 8];
#pragma unroll
                    for (int m = 0; m < 3; ++m) {
                        // bf16x2 split: hi*hi + hi*lo + lo*hi accumulated in f32
                        acc[t][m] = __builtin_amdgcn_wmma_f32_16x16x32_bf16(
                            false, ah.v, false, bhf[m], (short)0, acc[t][m], false, false);
                        acc[t][m] = __builtin_amdgcn_wmma_f32_16x16x32_bf16(
                            false, ah.v, false, blf[m], (short)0, acc[t][m], false, false);
                        acc[t][m] = __builtin_amdgcn_wmma_f32_16x16x32_bf16(
                            false, al.v, false, bhf[m], (short)0, acc[t][m], false, false);
                    }
                }
            }
            __syncthreads();   // all reads done before buffer is overwritten
        }

        // Gating + in-register affine-scan composition.
        // C/D layout: lanes 0-15 hold rows 0-7 (vgpr r = row r), lanes 16-31 rows 8-15.
#pragma unroll
        for (int t = 0; t < 4; ++t) {
            float Ath = 1.0f, Dth = 0.0f;
#pragma unroll
            for (int r = 0; r < 8; ++r) {
                float lam = softplus_f(acc[t][0][r] + blam);
                float del = softplus_f(acc[t][1][r] + bdel);
                float u   = acc[t][2][r] + bBv;
                float a   = expf(-del * lam);
                Ath = a * Ath;
                Dth = fmaf(a, Dth, del * u);
            }
            // merge the two half-tile affine maps (rows 0-7 then 8-15)
            float Ao = __shfl_xor(Ath, 16, 32);
            float Do = __shfl_xor(Dth, 16, 32);
            float Alo = hiHalf ? Ao  : Ath;
            float Dlo = hiHalf ? Do  : Dth;
            float Ahi = hiHalf ? Ath : Ao;
            float Dhi = hiHalf ? Dth : Do;
            float At = Ahi * Alo;
            float Dt = fmaf(Ahi, Dlo, Dhi);
            // compose tile map after carry: new = tile ∘ carry
            Dcar = fmaf(At, Dcar, Dt);
            Acar = At * Acar;
        }
    }

    if (!hiHalf) {
        int idx = (b * NCHUNK + chunk) * DMODEL + n;
        Achunk[idx] = Acar;
        Dchunk[idx] = Dcar;
    }
}

// ---------------- compose the chunk affine maps -> final state s ----------------
__global__ void compose_chunks_kernel(const float* __restrict__ Achunk,
                                      const float* __restrict__ Dchunk,
                                      float* __restrict__ s) {
    int i = blockIdx.x * blockDim.x + threadIdx.x;
    if (i >= B_SZ * DMODEL) return;
    int b = i >> 10;
    int d = i & 1023;
    float sv = 0.0f;
    for (int c = 0; c < NCHUNK; ++c) {
        int idx = (b * NCHUNK + c) * DMODEL + d;
        sv = fmaf(Achunk[idx], sv, Dchunk[idx]);
    }
    s[i] = sv;
}

// ---------------- output heads: parity (B,2) and add (B,1) ----------------
__global__ void heads_kernel(const float* __restrict__ s,
                             const float* __restrict__ W_par, const float* __restrict__ b_par,
                             const float* __restrict__ W_add, const float* __restrict__ b_add,
                             float* __restrict__ out) {
    int ob = blockIdx.x;        // 0..23
    int b = ob / 3;
    int j = ob % 3;             // 0,1 -> parity col, 2 -> add
    int tid = threadIdx.x;
    float p = 0.0f;
    for (int d = tid; d < DMODEL; d += 256) {
        float w = (j < 2) ? W_par[d * 2 + j] : W_add[d];
        p = fmaf(s[b * DMODEL + d], w, p);
    }
    __shared__ float red[256];
    red[tid] = p;
    __syncthreads();
    for (int off = 128; off > 0; off >>= 1) {
        if (tid < off) red[tid] += red[tid + off];
        __syncthreads();
    }
    if (tid == 0) {
        float v = red[0] + ((j < 2) ? b_par[j] : b_add[0]);
        if (j < 2) out[b * 2 + j] = v;      // parity_logits flat (8,2)
        else       out[16 + b]    = v;      // add_pred flat (8,1) after parity
    }
}

extern "C" void kernel_launch(void* const* d_in, const int* in_sizes, int n_in,
                              void* d_out, int out_size, void* d_ws, size_t ws_size,
                              hipStream_t stream) {
    const float* x     = (const float*)d_in[0];
    const float* W_lam = (const float*)d_in[1];
    const float* b_lam = (const float*)d_in[2];
    const float* W_del = (const float*)d_in[3];
    const float* b_del = (const float*)d_in[4];
    const float* W_B   = (const float*)d_in[5];
    const float* b_B   = (const float*)d_in[6];
    const float* W_par = (const float*)d_in[7];
    const float* b_par = (const float*)d_in[8];
    const float* W_add = (const float*)d_in[9];
    const float* b_add = (const float*)d_in[10];
    float* out = (float*)d_out;

    const size_t XN = (size_t)B_SZ * T_SZ * DIN;       // 33,554,432 elems
    const size_t WN = (size_t)3 * DIN * DMODEL;        //  3,145,728 elems

    bf16_t* xh = (bf16_t*)d_ws;
    bf16_t* xl = xh + XN;
    bf16_t* wh = xl + XN;
    bf16_t* wl = wh + WN;
    float* Achunk = (float*)(wl + WN);                  // B*NCHUNK*DMODEL
    float* Dchunk = Achunk + (size_t)B_SZ * NCHUNK * DMODEL;
    float* sbuf   = Dchunk + (size_t)B_SZ * NCHUNK * DMODEL;

    // 1) split x into bf16 hi/lo
    split_x_kernel<<<(int)(XN / 256), 256, 0, stream>>>(x, xh, xl, (int)XN);
    // 2) transpose + split the three projection weights
    split_w_kernel<<<(int)(WN / 256), 256, 0, stream>>>(W_lam, W_del, W_B, wh, wl);
    // 3) fused triple-GEMM + softplus/exp gating + chunk-local affine scan
    dim3 grid(DMODEL / BN, B_SZ, NCHUNK);
    mamba_proj_scan_kernel<<<grid, 256, 0, stream>>>(xh, xl, wh, wl,
                                                     b_lam, b_del, b_B,
                                                     Achunk, Dchunk);
    // 4) compose chunk maps -> final state
    compose_chunks_kernel<<<(B_SZ * DMODEL) / 256, 256, 0, stream>>>(Achunk, Dchunk, sbuf);
    // 5) heads
    heads_kernel<<<24, 256, 0, stream>>>(sbuf, W_par, b_par, W_add, b_add, out);
    (void)in_sizes; (void)n_in; (void)out_size; (void)ws_size;
}
